// PocketEncoderV3_11931419148542
// MI455X (gfx1250) — matrix operand order repaired and checked by reading, not compiled, not run on hardware
//
#include <hip/hip_runtime.h>
#include <math.h>

// ---------------------------------------------------------------------------
// Semla-style pocket encoder for MI455X (gfx1250, wave32, WMMA).
// All dense contractions run through v_wmma_f32_16x16x32_bf16 (bf16 in, f32
// accum). Weights are pre-swizzled into WMMA-fragment-native layout so every
// operand fragment is exactly 2 x b128 loads per lane; inner loops are
// branch-free with pointer-bumped addressing.
// ---------------------------------------------------------------------------

typedef __attribute__((ext_vector_type(16))) __bf16 v16bf;
typedef __attribute__((ext_vector_type(8)))  float  v8f;

union FragBF {
  v16bf v;
  unsigned int u[8];
  uint4 q[2];
};

__device__ inline v8f wmma_bf16(v16bf a, v16bf b, v8f c) {
  return __builtin_amdgcn_wmma_f32_16x16x32_bf16(false, a, false, b, (short)0, c,
                                                 false, false);
}

__device__ inline unsigned int pack2bf(float a, float b) {
  union { __bf16 h; unsigned short s; } x, y;
  x.h = (__bf16)a;
  y.h = (__bf16)b;
  return (unsigned int)x.s | ((unsigned int)y.s << 16);
}

// Load an A fragment from a per-lane row pointer (already offset by the
// lane's kb sub-offset): two contiguous 16B chunks at +0 and +16 bytes.
__device__ inline v16bf frag_a_ptr(const __bf16* r) {
  const uint4* p = reinterpret_cast<const uint4*>(r);
  FragBF f;
  f.q[0] = p[0];
  f.q[1] = p[1];
  return f.v;
}

// B fragment from swizzled storage: 32 contiguous bytes per lane.
__device__ inline v16bf frag_b_ptr(const unsigned int* w) {
  const uint4* p = reinterpret_cast<const uint4*>(w);
  FragBF f;
  f.q[0] = p[0];
  f.q[1] = p[1];
  return f.v;
}

// Gather fallback for non-swizzled (activation) B operands.
__device__ inline v16bf frag_b_gather(const __bf16* tile, int ld, int lane) {
  FragBF f;
  int n = lane & 15;
  int kb = (lane & 16) ? 16 : 0;
#pragma unroll
  for (int v2 = 0; v2 < 8; ++v2) {
    const __bf16* p0 = tile + (long)(kb + 2 * v2) * ld + n;
    unsigned int lo = *reinterpret_cast<const unsigned short*>(p0);
    unsigned int hi = *reinterpret_cast<const unsigned short*>(p0 + ld);
    f.u[v2] = lo | (hi << 16);
  }
  return f.v;
}

enum { F_BIAS = 1, F_SILU = 2, F_RESID = 4, F_MASKMUL = 8, F_NEGINF = 16 };

// ---------------------------------------------------------------------------
// Weight pre-swizzle: W[K,Nc] f32 (row-major, leading dim ldw) -> fragment
// layout [kt][ct][lane][8] u32, each u32 = two bf16 along K.
// ---------------------------------------------------------------------------
__global__ __launch_bounds__(256) void pack_b_kernel(
    const float* __restrict__ W, int ldw, int K, int Nc,
    unsigned int* __restrict__ out) {
  long idx = (long)blockIdx.x * 256 + threadIdx.x;
  long total = (long)K * Nc / 2;
  if (idx >= total) return;
  int v2 = (int)(idx & 7);
  int lane = (int)((idx >> 3) & 31);
  long t = idx >> 8;
  int nct = Nc >> 4;
  int ct = (int)(t % nct);
  int kt = (int)(t / nct);
  int n = ct * 16 + (lane & 15);
  int kb = (lane & 16) ? 16 : 0;
  int k = kt * 32 + kb + 2 * v2;
  out[idx] = pack2bf(W[(long)k * ldw + n], W[(long)(k + 1) * ldw + n]);
}

// ---------------------------------------------------------------------------
// Generic WMMA GEMM: out = epilogue(A[M,K]bf16 @ W[K,Nc]bf16).
// 256 threads = 8 waves arranged 4(row) x 2(col); wave computes a 32x32
// block (2x2 WMMA tiles): 4 WMMAs per k-step against 8 b128 loads.
// Block tile: 128(M) x 64(N). M % 128 == 0, Nc % 32 == 0, K % 32 == 0.
// ---------------------------------------------------------------------------
template <bool WSW>
__global__ __launch_bounds__(256) void gemm_bf16_kernel(
    const __bf16* __restrict__ A, int lda, long sA,
    const void* __restrict__ Wv_, int ldw, long sW,
    const float* __restrict__ bias,
    float* outf, long sOF,
    __bf16* outb, long sOB,
    const unsigned char* __restrict__ rowmask, int rm_div,
    int M, int K, int Nc, int flags) {
  int wave = threadIdx.x >> 5, lane = threadIdx.x & 31;
  int row0 = blockIdx.x * 128 + (wave >> 1) * 32;
  int col0 = blockIdx.y * 64 + (wave & 1) * 32;
  if (col0 >= Nc) return;
  A += (long)blockIdx.z * sA;
  if (outf) outf += (long)blockIdx.z * sOF;
  if (outb) outb += (long)blockIdx.z * sOB;
  int nct = Nc >> 4;

  const __bf16* a0 =
      A + ((long)row0 + (lane & 15)) * lda + ((lane & 16) ? 16 : 0);
  const __bf16* a1 = a0 + 16 * (long)lda;
  const unsigned int* wp = nullptr;
  const __bf16* wg = nullptr;
  if (WSW)
    wp = (const unsigned int*)Wv_ + (long)blockIdx.z * sW +
         ((long)(col0 >> 4) * 32 + lane) * 8;
  else
    wg = (const __bf16*)Wv_ + (long)blockIdx.z * sW;

  v8f acc[2][2] = {};
  for (int k0 = 0; k0 < K; k0 += 32) {
    v16bf fa0 = frag_a_ptr(a0);
    v16bf fa1 = frag_a_ptr(a1);
    v16bf fb0, fb1;
    if (WSW) {
      fb0 = frag_b_ptr(wp);
      fb1 = frag_b_ptr(wp + 256);  // next 16-col tile
      wp += (long)nct * 256;
    } else {
      fb0 = frag_b_gather(wg + (long)k0 * ldw + col0, ldw, lane);
      fb1 = frag_b_gather(wg + (long)k0 * ldw + col0 + 16, ldw, lane);
    }
    acc[0][0] = wmma_bf16(fa0, fb0, acc[0][0]);
    acc[0][1] = wmma_bf16(fa0, fb1, acc[0][1]);
    acc[1][0] = wmma_bf16(fa1, fb0, acc[1][0]);
    acc[1][1] = wmma_bf16(fa1, fb1, acc[1][1]);
    a0 += 32;
    a1 += 32;
  }

#pragma unroll
  for (int rt = 0; rt < 2; ++rt) {
#pragma unroll
    for (int ct = 0; ct < 2; ++ct) {
      int col = col0 + ct * 16 + (lane & 15);
      int rbase = row0 + rt * 16 + ((lane & 16) ? 8 : 0);
      float bval = (flags & F_BIAS) ? bias[col] : 0.f;
#pragma unroll
      for (int r = 0; r < 8; ++r) {
        int row = rbase + r;
        float v = acc[rt][ct][r] + bval;
        if (flags & F_SILU) v = v / (1.f + expf(-v));
        float m = 1.f;
        if (rowmask) m = (float)rowmask[row / rm_div];
        if (flags & F_NEGINF) { if (m == 0.f) v = -3.4028235e38f; }
        else if (flags & F_MASKMUL) v *= m;
        long idx = (long)row * Nc + col;
        if (flags & F_RESID) outf[idx] += v * m;
        else if (outf) outf[idx] = v;
        if (outb) outb[idx] = (__bf16)v;
      }
    }
  }
}

// ---------------------------------------------------------------------------
// Pairwise message hidden layer, one block per (b,i):
//   h[j,f] = kb[j,0:192] @ W'_i[192,128] + edge[j,0:128] @ We[128,128]
//            + a_part[i,f] + b_part[j,f] + b1[f];  hs = silu(h) (bf16)
// W'_i[(x*64+m), f] = qa[i,x,m] * Wd[m,f], built in swizzled fragment layout
// in LDS (48 KB). Two branch-free phases: K=192 (LDS B) then K=128 (We).
// MODE 0: edge rows from x_edge; MODE 1: gather via adjacency (hoisted).
// ---------------------------------------------------------------------------
template <int MODE>
__global__ __launch_bounds__(256) void pairwise_h_kernel(
    const float* __restrict__ qa,        // [B*N, 3*64] f32
    const __bf16* __restrict__ kbb,      // [B*N, 192] bf16
    const float* __restrict__ Wd,        // [64,128] f32
    const unsigned int* __restrict__ Wedge_sw,  // swizzled [128,128]
    const __bf16* __restrict__ edge,     // MODE0: [B*N*N,128]; MODE1: bond tab [5,128]
    const int* __restrict__ adjacency,   // MODE1 only: [B*N*N]
    const float* __restrict__ a_part,    // [B*N,128]
    const float* __restrict__ b_part,    // [B*N,128]
    const float* __restrict__ b1,        // [128]
    __bf16* __restrict__ hs) {           // [B*N*N,128]
  __shared__ unsigned int Wp[6 * 8 * 32 * 8];  // 12288 u32 = 48 KB
  int bi = blockIdx.x;  // b*256 + i
  int b = bi >> 8;
  long qbase = (long)bi * 192;
  // Build W'_i directly in swizzled fragment layout: idx -> (kt,ct,lane,v2).
  for (int t = threadIdx.x; t < 12288; t += 256) {
    int v2 = t & 7;
    int ln = (t >> 3) & 31;
    int ct = (t >> 8) & 7;
    int kt = t >> 11;  // 0..5
    int f = ct * 16 + (ln & 15);
    int kb = (ln & 16) ? 16 : 0;
    int k0 = kt * 32 + kb + 2 * v2;  // xm index, < 192
    int k1 = k0 + 1;
    float w0 = qa[qbase + (k0 >> 6) * 64 + (k0 & 63)] * Wd[(k0 & 63) * 128 + f];
    float w1 = qa[qbase + (k1 >> 6) * 64 + (k1 & 63)] * Wd[(k1 & 63) * 128 + f];
    Wp[t] = pack2bf(w0, w1);
  }
  __syncthreads();

  int wave = threadIdx.x >> 5, lane = threadIdx.x & 31;
  int j0 = wave * 32;
  int kb8 = (lane & 16) ? 16 : 0;
  v8f acc[2][8] = {};

  // ---- phase 1: K = 192, A rows from kbb, B from LDS W'_i ----
  {
    const __bf16* ar0 =
        kbb + ((long)b * 256 + j0 + (lane & 15)) * 192 + kb8;
    const __bf16* ar1 = ar0 + 16 * 192;
    const unsigned int* wl = &Wp[lane * 8];
    for (int kt = 0; kt < 6; ++kt) {
      v16bf fa0 = frag_a_ptr(ar0);
      v16bf fa1 = frag_a_ptr(ar1);
#pragma unroll
      for (int ct = 0; ct < 8; ++ct) {
        v16bf fb = frag_b_ptr(wl + ct * 256);
        acc[0][ct] = wmma_bf16(fa0, fb, acc[0][ct]);
        acc[1][ct] = wmma_bf16(fa1, fb, acc[1][ct]);
      }
      ar0 += 32;
      ar1 += 32;
      wl += 8 * 256;
    }
  }
  // ---- phase 2: K = 128, A rows from edge source, B from swizzled We ----
  {
    const __bf16* er0;
    const __bf16* er1;
    if (MODE == 0) {
      er0 = edge + ((long)bi * 256 + j0 + (lane & 15)) * 128 + kb8;
      er1 = er0 + 16 * 128;
    } else {
      int a0i = adjacency[(long)bi * 256 + j0 + (lane & 15)];
      int a1i = adjacency[(long)bi * 256 + j0 + 16 + (lane & 15)];
      er0 = edge + (long)a0i * 128 + kb8;
      er1 = edge + (long)a1i * 128 + kb8;
    }
    const unsigned int* wl = Wedge_sw + lane * 8;
    for (int kt = 0; kt < 4; ++kt) {
      v16bf fa0 = frag_a_ptr(er0);
      v16bf fa1 = frag_a_ptr(er1);
#pragma unroll
      for (int ct = 0; ct < 8; ++ct) {
        v16bf fb = frag_b_ptr(wl + ct * 256);
        acc[0][ct] = wmma_bf16(fa0, fb, acc[0][ct]);
        acc[1][ct] = wmma_bf16(fa1, fb, acc[1][ct]);
      }
      er0 += 32;
      er1 += 32;
      wl += 8 * 256;
    }
  }

  const float* ap = a_part + (long)bi * 128;
#pragma unroll
  for (int rt = 0; rt < 2; ++rt) {
    int jb = j0 + rt * 16 + ((lane & 16) ? 8 : 0);
#pragma unroll
    for (int ct = 0; ct < 8; ++ct) {
      int f = ct * 16 + (lane & 15);
      float add = ap[f] + b1[f];
#pragma unroll
      for (int r = 0; r < 8; ++r) {
        int j = jb + r;
        float v = acc[rt][ct][r] + add + b_part[((long)b * 256 + j) * 128 + f];
        v = v / (1.f + expf(-v));
        hs[((long)bi * 256 + j) * 128 + f] = (__bf16)v;
      }
    }
  }
}

// ---------------------------------------------------------------------------
// Small VALU helper kernels
// ---------------------------------------------------------------------------
__global__ void f2b_kernel(const float* __restrict__ in,
                           __bf16* __restrict__ out, long n) {
  long i = (long)blockIdx.x * blockDim.x + threadIdx.x;
  if (i < n) out[i] = (__bf16)in[i];
}

__global__ __launch_bounds__(384) void embed_kernel(
    const float* __restrict__ coords, const int* __restrict__ atoms,
    const int* __restrict__ charges, const int* __restrict__ residues,
    const float* __restrict__ atom_emb, const float* __restrict__ charge_emb,
    const float* __restrict__ res_emb, const float* __restrict__ coord_W,
    __bf16* __restrict__ inv_in, float* __restrict__ x_equi) {
  int node = blockIdx.x, t = threadIdx.x;
  int a = atoms[node], c = charges[node], rr = residues[node];
  inv_in[(long)node * 1152 + t] = (__bf16)atom_emb[a * 384 + t];
  inv_in[(long)node * 1152 + 384 + t] = (__bf16)charge_emb[c * 384 + t];
  inv_in[(long)node * 1152 + 768 + t] = (__bf16)res_emb[rr * 384 + t];
  if (t < 288) {
    int x = t / 96, e = t % 96;
    x_equi[(long)node * 288 + t] = coords[node * 3 + x] * coord_W[e];
  }
}

__global__ __launch_bounds__(256) void pairmask_kernel(
    const unsigned char* __restrict__ mask, unsigned char* __restrict__ pm) {
  int bi = blockIdx.x, j = threadIdx.x;
  int b = bi >> 8;
  pm[(long)bi * 256 + j] = (unsigned char)(mask[bi] & mask[b * 256 + j]);
}

__global__ __launch_bounds__(128) void layernorm_kernel(
    const float* __restrict__ x, const float* __restrict__ g,
    const float* __restrict__ be, __bf16* __restrict__ out) {
  __shared__ float red[128];
  int node = blockIdx.x, t = threadIdx.x;
  const float* xr = x + (long)node * 384;
  float s = 0.f;
  for (int i = t; i < 384; i += 128) s += xr[i];
  red[t] = s; __syncthreads();
  for (int o = 64; o > 0; o >>= 1) { if (t < o) red[t] += red[t + o]; __syncthreads(); }
  float mu = red[0] / 384.f; __syncthreads();
  float s2 = 0.f;
  for (int i = t; i < 384; i += 128) { float d = xr[i] - mu; s2 += d * d; }
  red[t] = s2; __syncthreads();
  for (int o = 64; o > 0; o >>= 1) { if (t < o) red[t] += red[t + o]; __syncthreads(); }
  float rs = rsqrtf(red[0] / 384.f + 1e-5f);
  for (int i = t; i < 384; i += 128)
    out[(long)node * 384 + i] = (__bf16)((xr[i] - mu) * rs * g[i] + be[i]);
}

__global__ __launch_bounds__(256) void coordnorm_stats_kernel(
    const float* __restrict__ xe, const unsigned char* __restrict__ mask,
    float* __restrict__ meanlen) {
  __shared__ float rl[256], rc[256];
  int b = blockIdx.x / 96, e = blockIdx.x % 96;
  int n = threadIdx.x;
  long base = (long)(b * 256 + n) * 288 + e;
  float x0 = xe[base], x1 = xe[base + 96], x2 = xe[base + 192];
  float m = (float)mask[b * 256 + n];
  rl[n] = sqrtf(x0 * x0 + x1 * x1 + x2 * x2 + 1e-12f) * m;
  rc[n] = m;
  __syncthreads();
  for (int o = 128; o > 0; o >>= 1) {
    if (n < o) { rl[n] += rl[n + o]; rc[n] += rc[n + o]; }
    __syncthreads();
  }
  if (n == 0) meanlen[b * 96 + e] = rl[0] / fmaxf(rc[0], 1.f);
}

__global__ __launch_bounds__(256) void coordnorm_apply_kernel(
    const float* __restrict__ xe, const float* __restrict__ meanlen,
    const float* __restrict__ scale, const unsigned char* __restrict__ mask,
    float* __restrict__ eqn, __bf16* __restrict__ eqnb) {
  long i = (long)blockIdx.x * 256 + threadIdx.x;  // total B*N*3*96 = 294912
  int e = (int)(i % 96);
  long node = i / 288;
  int b = (int)(node >> 8);
  float m = (float)mask[node];
  float v = xe[i] * scale[e] / (meanlen[b * 96 + e] + 1e-3f) * m;
  eqn[i] = v;
  eqnb[i] = (__bf16)v;
}

__global__ __launch_bounds__(64) void softmax_kernel(
    const float* __restrict__ logits, __bf16* __restrict__ attn) {
  int bi = blockIdx.x, h = threadIdx.x;
  const float* lr = logits + (long)bi * 256 * 64 + h;
  float mx = -3.4028235e38f;
  for (int j = 0; j < 256; ++j) mx = fmaxf(mx, lr[(long)j * 64]);
  float s = 0.f;
  for (int j = 0; j < 256; ++j) s += expf(lr[(long)j * 64] - mx);
  float inv = 1.f / s;
  __bf16* ar = attn + (long)bi * 256 * 64 + h;
  for (int j = 0; j < 256; ++j)
    ar[(long)j * 64] = (__bf16)(expf(lr[(long)j * 64] - mx) * inv);
}

__global__ __launch_bounds__(256) void attnmean_kernel(
    const __bf16* __restrict__ attn, __bf16* __restrict__ wmean) {
  int bi = blockIdx.x, j = threadIdx.x;
  const __bf16* ar = attn + ((long)bi * 256 + j) * 64;
  float s = 0.f;
  for (int h = 0; h < 64; ++h) s += (float)ar[h];
  wmean[(long)bi * 256 + j] = (__bf16)(s * (1.f / 64.f));
}

__global__ __launch_bounds__(384) void headagg_kernel(
    const __bf16* __restrict__ attn, const float* __restrict__ v,
    __bf16* __restrict__ agg) {
  int bi = blockIdx.x, t = threadIdx.x;  // t = h*6+d
  int b = bi >> 8;
  int h = t / 6;
  float s = 0.f;
  for (int j = 0; j < 256; ++j)
    s += (float)attn[((long)bi * 256 + j) * 64 + h] *
         v[(long)(b * 256 + j) * 384 + t];
  agg[(long)bi * 384 + t] = (__bf16)s;
}

// ---------------------------------------------------------------------------
// Host side
// ---------------------------------------------------------------------------
static void gemm(hipStream_t s, const __bf16* A, int lda, long sA,
                 const void* W, int ldw, long sW, int wsw, const float* bias,
                 float* outf, long sOF, __bf16* outb, long sOB,
                 const unsigned char* rm, int rmd, int M, int K, int Nc,
                 int flags, int batch = 1) {
  dim3 g((M + 127) / 128, (Nc + 63) / 64, batch);
  if (wsw)
    gemm_bf16_kernel<true><<<g, 256, 0, s>>>(A, lda, sA, W, ldw, sW, bias,
                                             outf, sOF, outb, sOB, rm, rmd, M,
                                             K, Nc, flags);
  else
    gemm_bf16_kernel<false><<<g, 256, 0, s>>>(A, lda, sA, W, ldw, sW, bias,
                                              outf, sOF, outb, sOB, rm, rmd, M,
                                              K, Nc, flags);
}

extern "C" void kernel_launch(void* const* d_in, const int* in_sizes, int n_in,
                              void* d_out, int out_size, void* d_ws,
                              size_t ws_size, hipStream_t stream) {
  (void)in_sizes; (void)n_in; (void)out_size; (void)ws_size;
  const float* coords = (const float*)d_in[0];
  const int* atoms = (const int*)d_in[1];
  const int* charges = (const int*)d_in[2];
  const int* residues = (const int*)d_in[3];
  const int* adjacency = (const int*)d_in[4];
  const unsigned char* mask = (const unsigned char*)d_in[5];
  auto P = [&](int i) { return (const float*)d_in[i]; };

  float* xequi = (float*)d_out;          // [B,N,3,96]
  float* xinv = (float*)d_out + 294912;  // [B,N,384]

  char* ws = (char*)d_ws;
  size_t cur = 0;
  auto alloc = [&](size_t bytes) -> char* {
    char* p = ws + cur;
    cur = (cur + bytes + 255) & ~(size_t)255;
    return p;
  };

  __bf16* invin = (__bf16*)alloc(1024L * 1152 * 2);
  __bf16* xequib = (__bf16*)alloc(294912L * 2);
  __bf16* xinvb = (__bf16*)alloc(393216L * 2);
  float* eqn = (float*)alloc(294912L * 4);
  __bf16* eqnb = (__bf16*)alloc(294912L * 2);
  __bf16* invnb = (__bf16*)alloc(393216L * 2);
  float* qaf = (float*)alloc(3072L * 64 * 4);
  __bf16* kbbuf = (__bf16*)alloc(3072L * 64 * 2);
  float* apart = (float*)alloc(1024L * 128 * 4);
  float* bpart = (float*)alloc(1024L * 128 * 4);
  float* mlen = (float*)alloc(4L * 96 * 4);
  unsigned char* pm = (unsigned char*)alloc(262144);
  __bf16* hs = (__bf16*)alloc(262144L * 128 * 2);  // aliased as attn later
  float* logits = (float*)alloc(262144L * 64 * 4);
  __bf16* attn = hs;  // hs is dead once logits are computed; reuse as attn
  __bf16* wmean = (__bf16*)alloc(262144L * 2);
  __bf16* xedgeb = (__bf16*)alloc(262144L * 128 * 2);
  float* vbuf = (float*)alloc(1024L * 384 * 4);
  __bf16* agg1 = (__bf16*)alloc(1024L * 384 * 2);
  __bf16* equivb = (__bf16*)alloc(3072L * 96 * 2);
  __bf16* equiaggb = (__bf16*)alloc(3072L * 96 * 2);
  __bf16* inv2b = (__bf16*)alloc(393216L * 2);
  __bf16* ff1b = (__bf16*)alloc(1024L * 768 * 2);

  auto conv = [&](const float* src, long n) -> const __bf16* {
    __bf16* dst = (__bf16*)alloc(n * 2);
    f2b_kernel<<<(unsigned)((n + 255) / 256), 256, 0, stream>>>(src, dst, n);
    return dst;
  };
  auto packW = [&](const float* W, int ldw, int K, int Nc) -> const unsigned int* {
    unsigned int* dst = (unsigned int*)alloc((size_t)K * Nc * 2);
    long total = (long)K * Nc / 2;
    pack_b_kernel<<<(unsigned)((total + 255) / 256), 256, 0, stream>>>(W, ldw, K,
                                                                      Nc, dst);
    return dst;
  };

  const unsigned int* invWp = packW(P(10), 384, 1152, 384);
  const __bf16* bondtabb = conv(P(9), 5L * 128);
  const unsigned int* b_aprojp = packW(P(13), 64, 96, 64);
  const unsigned int* b_bprojp = packW(P(14), 64, 96, 64);
  const unsigned int* b_Wap = packW(P(15), 128, 384, 128);
  const unsigned int* b_Wbp = packW(P(15) + 384 * 128, 128, 384, 128);
  const unsigned int* b_Wedgep = packW(P(15) + 832 * 128, 128, 128, 128);
  const unsigned int* b_W2p = packW(P(17), 128, 128, 128);
  struct LW {
    const unsigned int *aproj, *bproj, *Wa, *Wb, *Wedge, *W2, *Wv, *Wo, *We,
        *Weo, *Wf1, *Wf2;
  } lw[4];
  for (int l = 0; l < 4; ++l) {
    int base = 19 + l * 19;
    lw[l].aproj = packW(P(base + 3), 64, 96, 64);
    lw[l].bproj = packW(P(base + 4), 64, 96, 64);
    lw[l].Wa = packW(P(base + 5), 128, 384, 128);
    lw[l].Wb = packW(P(base + 5) + 384 * 128, 128, 384, 128);
    lw[l].Wedge = packW(P(base + 5) + 832 * 128, 128, 128, 128);
    lw[l].W2 = packW(P(base + 7), 64, 128, 64);
    lw[l].Wv = packW(P(base + 9), 384, 384, 384);
    lw[l].Wo = packW(P(base + 10), 384, 384, 384);
    lw[l].We = packW(P(base + 11), 96, 96, 96);
    lw[l].Weo = packW(P(base + 12), 96, 96, 96);
    lw[l].Wf1 = packW(P(base + 15), 768, 384, 768);
    lw[l].Wf2 = packW(P(base + 17), 384, 768, 384);
  }

  // ---- embeddings ----
  embed_kernel<<<1024, 384, 0, stream>>>(coords, atoms, charges, residues,
                                         P(7), P(8), P(6), P(12), invin, xequi);
  pairmask_kernel<<<1024, 256, 0, stream>>>(mask, pm);
  // x_inv = silu(inv_in @ inv_W + inv_b) * mask
  gemm(stream, invin, 1152, 0, invWp, 0, 0, 1, P(11), xinv, 0, nullptr, 0,
       mask, 1, 1024, 1152, 384, F_BIAS | F_SILU | F_MASKMUL);
  f2b_kernel<<<(294912 + 255) / 256, 256, 0, stream>>>(xequi, xequib, 294912);
  f2b_kernel<<<(393216 + 255) / 256, 256, 0, stream>>>(xinv, xinvb, 393216);

  // ---- bond pairwise MLP -> x_edge ----
  gemm(stream, xequib, 96, 0, b_aprojp, 0, 0, 1, nullptr, qaf, 0, nullptr, 0,
       nullptr, 1, 3072, 96, 64, 0);
  gemm(stream, xequib, 96, 0, b_bprojp, 0, 0, 1, nullptr, nullptr, 0, kbbuf, 0,
       nullptr, 1, 3072, 96, 64, 0);
  gemm(stream, xinvb, 384, 0, b_Wap, 0, 0, 1, nullptr, apart, 0, nullptr, 0,
       nullptr, 1, 1024, 384, 128, 0);
  gemm(stream, xinvb, 384, 0, b_Wbp, 0, 0, 1, nullptr, bpart, 0, nullptr, 0,
       nullptr, 1, 1024, 384, 128, 0);
  pairwise_h_kernel<1><<<1024, 256, 0, stream>>>(
      qaf, kbbuf, P(15) + 768 * 128, b_Wedgep, bondtabb, adjacency, apart,
      bpart, P(16), hs);
  gemm(stream, hs, 128, 0, b_W2p, 0, 0, 1, P(18), nullptr, 0, xedgeb, 0, pm, 1,
       262144, 128, 128, F_BIAS | F_MASKMUL);

  // ---- layers ----
  for (int l = 0; l < 4; ++l) {
    int base = 19 + l * 19;
    coordnorm_stats_kernel<<<384, 256, 0, stream>>>(xequi, mask, mlen);
    coordnorm_apply_kernel<<<1152, 256, 0, stream>>>(xequi, mlen, P(base + 0),
                                                     mask, eqn, eqnb);
    layernorm_kernel<<<1024, 128, 0, stream>>>(xinv, P(base + 1), P(base + 2),
                                               invnb);
    gemm(stream, eqnb, 96, 0, lw[l].aproj, 0, 0, 1, nullptr, qaf, 0, nullptr,
         0, nullptr, 1, 3072, 96, 64, 0);
    gemm(stream, eqnb, 96, 0, lw[l].bproj, 0, 0, 1, nullptr, nullptr, 0, kbbuf,
         0, nullptr, 1, 3072, 96, 64, 0);
    gemm(stream, invnb, 384, 0, lw[l].Wa, 0, 0, 1, nullptr, apart, 0, nullptr,
         0, nullptr, 1, 1024, 384, 128, 0);
    gemm(stream, invnb, 384, 0, lw[l].Wb, 0, 0, 1, nullptr, bpart, 0, nullptr,
         0, nullptr, 1, 1024, 384, 128, 0);
    pairwise_h_kernel<0><<<1024, 256, 0, stream>>>(
        qaf, kbbuf, P(base + 5) + 768 * 128, lw[l].Wedge, xedgeb, nullptr,
        apart, bpart, P(base + 6), hs);
    gemm(stream, hs, 128, 0, lw[l].W2, 0, 0, 1, P(base + 8), logits, 0,
         nullptr, 0, pm, 1, 262144, 128, 64, F_BIAS | F_NEGINF);
    softmax_kernel<<<1024, 64, 0, stream>>>(logits, attn);
    attnmean_kernel<<<1024, 256, 0, stream>>>(attn, wmean);
    gemm(stream, invnb, 384, 0, lw[l].Wv, 0, 0, 1, nullptr, vbuf, 0, nullptr,
         0, nullptr, 1, 1024, 384, 384, 0);
    headagg_kernel<<<1024, 384, 0, stream>>>(attn, vbuf, agg1);
    gemm(stream, agg1, 384, 0, lw[l].Wo, 0, 0, 1, nullptr, xinv, 0, nullptr, 0,
         mask, 1, 1024, 384, 384, F_RESID);
    // equivariant path: equi_v = equi_n @ We ; equi_agg = wmean @ equi_v
    gemm(stream, eqnb, 96, 0, lw[l].We, 0, 0, 1, nullptr, nullptr, 0, equivb,
         0, nullptr, 1, 3072, 96, 96, 0);
    gemm(stream, wmean, 256, 65536, equivb, 288, 73728, 0, nullptr, nullptr, 0,
         equiaggb, 73728, nullptr, 1, 256, 256, 288, 0, 4);
    gemm(stream, equiaggb, 96, 0, lw[l].Weo, 0, 0, 1, nullptr, xequi, 0,
         nullptr, 0, mask, 3, 3072, 96, 96, F_RESID);
    // feedforward
    layernorm_kernel<<<1024, 128, 0, stream>>>(xinv, P(base + 13),
                                               P(base + 14), inv2b);
    gemm(stream, inv2b, 384, 0, lw[l].Wf1, 0, 0, 1, P(base + 16), nullptr, 0,
         ff1b, 0, nullptr, 1, 1024, 384, 768, F_BIAS | F_SILU);
    gemm(stream, ff1b, 768, 0, lw[l].Wf2, 0, 0, 1, P(base + 18), xinv, 0,
         nullptr, 0, mask, 1, 1024, 768, 384, F_BIAS | F_RESID);
  }
}
